// MinkPillarUNet_41798621725230
// MI455X (gfx1250) — matrix operand order repaired and checked by reading, not compiled, not run on hardware
//
#include <hip/hip_runtime.h>
#include <cstddef>
#include <cstdint>

// ============================================================================
// MinkPillarUNet forward for MI455X (gfx1250, wave32, WMMA).
//
// Core op: out[N,Cout] = sum_k gather(F via kmap[k]) @ W[k]   (sparse conv)
//
// Data path (gather-bandwidth bound; activations live in the 192MB L2):
//   - activations pre-converted once per GEMM to bf16, padded to 32 cols
//   - gathered A tile staged into LDS with GLOBAL_LOAD_ASYNC_TO_LDS_B32
//     (ASYNCcnt path, no VGPR roundtrip), shared across up to 4 waves
//   - weights pre-swizzled into exact WMMA B-fragment order
//     -> B fragments are two coalesced global_load_b128 per wave, no LDS
//   - v_wmma_f32_16x16x32_bf16, f32 accumulation
// ============================================================================

#define BLK 256
#define NWMAX 4

typedef __attribute__((ext_vector_type(16))) __bf16 bf16x16;
typedef __attribute__((ext_vector_type(8)))  float  f32x8;

__device__ __forceinline__ unsigned short f2bf(float f) {
  // round-to-nearest-even f32 -> bf16
  unsigned u = __float_as_uint(f);
  u += 0x7fffu + ((u >> 16) & 1u);
  return (unsigned short)(u >> 16);
}

union BFrag { bf16x16 v; uint4 q[2]; };

// ---------------------------------------------------------------------------
// Activations f32[N][Cin] -> bf16[N][Cp], Cp = Cin rounded up to 32 (zero pad)
// ---------------------------------------------------------------------------
__global__ __launch_bounds__(BLK)
void xtobf_kernel(const float* __restrict__ X, unsigned short* __restrict__ Xh,
                  int N, int Cin, int Cp)
{
  size_t i = (size_t)blockIdx.x * BLK + threadIdx.x;
  if (i >= (size_t)N * Cp) return;
  int r = (int)(i / Cp), c = (int)(i % Cp);
  Xh[i] = (c < Cin) ? f2bf(X[(size_t)r * Cin + c]) : (unsigned short)0;
}

// ---------------------------------------------------------------------------
// Weights f32[K][Cin][Cout] -> bf16 fragment order:
// Wf[(((k*NTp + nt)*CT + ct)*32 + lane)*16 + t] = W[k][ct*32 + kk][nt*16 + n]
//   n = lane&15, kk = (lane<16 ? t : 16+t)      (contiguous k = 16h+t, same order as the A fragment)
// Zero-padded for ci>=Cin, co>=Cout, nt beyond real column tiles.
// ---------------------------------------------------------------------------
__global__ __launch_bounds__(BLK)
void wtobf_kernel(const float* __restrict__ W, unsigned short* __restrict__ Wf,
                  int K, int Cin, int Cout, int CT, int NTp)
{
  size_t i = (size_t)blockIdx.x * BLK + threadIdx.x;
  size_t total = (size_t)K * NTp * CT * 512;
  if (i >= total) return;
  int t    = (int)(i & 15);
  int lane = (int)((i >> 4) & 31);
  size_t tile = i >> 9;
  int ct = (int)(tile % CT); size_t tmp = tile / CT;
  int nt = (int)(tmp % NTp); int k = (int)(tmp / NTp);
  int n  = lane & 15;
  int kk = (lane < 16) ? t : (16 + t);
  int ci = (ct << 5) + kk;
  int co = nt * 16 + n;
  float v = (ci < Cin && co < Cout) ? W[((size_t)k * Cin + ci) * Cout + co] : 0.f;
  Wf[i] = f2bf(v);
}

// ---------------------------------------------------------------------------
// WMMA gather-GEMM. Xh: bf16 activations [Nf][Cp]; Wf: fragment-order weights.
// kmap==nullptr -> identity gather (dense). kmap entries < 0 -> zero rows.
// Block: nw waves; wave w owns column tile nt = blockIdx.y*nw + w.
// ---------------------------------------------------------------------------
__global__ __launch_bounds__(32 * NWMAX)
void wmma_gemm_gather(const unsigned short* __restrict__ Xh, int Cp,
                      const unsigned short* __restrict__ Wf,
                      int K, const int* __restrict__ kmap, int Nout, int Cout,
                      const float* __restrict__ bias, float* __restrict__ Out,
                      int CT, int NTp)
{
  const int tid  = threadIdx.x;
  const int lane = tid & 31;
  const int wave = tid >> 5;
  const int nw   = blockDim.x >> 5;
  const int m0 = blockIdx.x * 16;                // output row tile
  const int nt = blockIdx.y * nw + wave;         // this wave's col tile
  const int n0 = nt * 16;

  __shared__ __align__(16) unsigned int As[256]; // 16 rows x 32 bf16 (shared A)

  f32x8 acc = {0.f, 0.f, 0.f, 0.f, 0.f, 0.f, 0.f, 0.f};

  for (int k = 0; k < K; ++k) {
    const int* kb = kmap ? (kmap + (size_t)k * Nout) : (const int*)0;
    for (int ct = 0; ct < CT; ++ct) {
      // ---- stage shared A tile: async gather straight into LDS ----
      for (int e = tid; e < 256; e += blockDim.x) {
        int r = e >> 4, c2 = (e & 15) << 1;
        int row = m0 + r;
        int idx = -1;
        if (row < Nout) idx = kb ? kb[row] : row;
        if (idx >= 0) {
          const unsigned short* gp = &Xh[(size_t)idx * Cp + (ct << 5) + c2];
          unsigned lds = (unsigned)(uintptr_t)&As[e];  // LDS offset = addr[31:0]
          asm volatile("global_load_async_to_lds_b32 %0, %1, off"
                       :: "v"(lds), "v"(gp) : "memory");
        } else {
          As[e] = 0u;  // zero-fill masked rows (DScnt path)
        }
      }
      // ---- B fragment: direct coalesced global loads, no LDS ----
      size_t tb = (((size_t)k * NTp + nt) * CT + ct) * 512;   // u16 elements
      if (ct + 1 < CT)  // prefetch next B tile (global_prefetch_b8)
        __builtin_prefetch(&Wf[tb + 512 + (size_t)lane * 32], 0, 1);
      BFrag b;
      b.q[0] = *(const uint4*)&Wf[tb + (size_t)lane * 16];
      b.q[1] = *(const uint4*)&Wf[tb + (size_t)lane * 16 + 8];

      asm volatile("s_wait_asynccnt 0" ::: "memory");  // async tile landed
      __syncthreads();                                 // + DScnt, all waves

      // ---- A fragment, contiguous k = 16h + s as in Wf (16-bit A 16x32, wave32) ----
      BFrag a;
      int ad = (lane & 15) * 16 + ((lane < 16) ? 0 : 8);  // dword index
      a.q[0] = *(const uint4*)&As[ad];
      a.q[1] = *(const uint4*)&As[ad + 4];

      acc = __builtin_amdgcn_wmma_f32_16x16x32_bf16(false, a.v, false, b.v,
                                                    (short)0, acc, false, false);
      __syncthreads();
    }
  }

  // ---- store C/D: VGPR j -> M = j (lanes 0-15) / j+8 (lanes 16-31) ----
  int col = n0 + (lane & 15);
  if (col < Cout) {
    float badd = bias ? bias[col] : 0.f;
    int rbase = m0 + ((lane < 16) ? 0 : 8);
#pragma unroll
    for (int j = 0; j < 8; ++j) {
      int row = rbase + j;
      if (row < Nout) Out[(size_t)row * Cout + col] = acc[j] + badd;
    }
  }
}

// ---------------------------------------------------------------------------
// Column statistics: stat[c]=sum, stat[C+c]=sumsq, stat[2C]=row count (masked).
// ---------------------------------------------------------------------------
__global__ __launch_bounds__(BLK)
void colstat_kernel(const float* __restrict__ X, int N, int C,
                    const float* __restrict__ mask, float* __restrict__ stat)
{
  __shared__ float sh[3 * BLK];
  int c = blockIdx.x, t = threadIdx.x;
  float s = 0.f, s2 = 0.f, cw = 0.f;
  for (int i = t; i < N; i += BLK) {
    float w = mask ? (mask[i] > 0.f ? 1.f : 0.f) : 1.f;
    float x = X[(size_t)i * C + c];
    s += w * x; s2 += w * x * x; cw += w;
  }
  sh[t] = s; sh[BLK + t] = s2; sh[2 * BLK + t] = cw;
  __syncthreads();
  for (int o = BLK / 2; o > 0; o >>= 1) {
    if (t < o) {
      sh[t] += sh[t + o];
      sh[BLK + t] += sh[BLK + t + o];
      sh[2 * BLK + t] += sh[2 * BLK + t + o];
    }
    __syncthreads();
  }
  if (t == 0) {
    stat[c] = sh[0];
    stat[C + c] = sh[BLK];
    if (c == 0) stat[2 * C] = mask ? sh[2 * BLK] : (float)N;
  }
}

// y = (x-m)*g*rsqrt(v+eps)+b  [+res]  [relu]
__global__ __launch_bounds__(BLK)
void bn_apply_kernel(const float* __restrict__ X, float* __restrict__ Y,
                     int N, int C, const float* __restrict__ stat,
                     const float* __restrict__ g, const float* __restrict__ b,
                     const float* __restrict__ res, int relu)
{
  size_t i = (size_t)blockIdx.x * BLK + threadIdx.x;
  if (i >= (size_t)N * C) return;
  int c = (int)(i % C);
  float P = stat[2 * C]; P = (P > 0.f) ? P : 1.f;
  float m = stat[c] / P;
  float v = stat[C + c] / P - m * m;
  float y = (X[i] - m) * (g[c] * rsqrtf(v + 1e-5f)) + b[c];
  if (res) y += res[i];
  if (relu) y = fmaxf(y, 0.f);
  Y[i] = y;
}

__global__ __launch_bounds__(BLK)
void zero_kernel(float* __restrict__ p, size_t n)
{
  size_t i = (size_t)blockIdx.x * BLK + threadIdx.x;
  if (i < n) p[i] = 0.f;
}

__global__ __launch_bounds__(BLK)
void seg_accum_kernel(const float* __restrict__ F, const int* __restrict__ seg,
                      int N, int C, float* __restrict__ psum, float* __restrict__ pcnt)
{
  size_t i = (size_t)blockIdx.x * BLK + threadIdx.x;
  if (i >= (size_t)N * C) return;
  int r = (int)(i / C), c = (int)(i % C);
  int p = seg[r];
  atomicAdd(&psum[(size_t)p * C + c], F[i]);
  if (c == 0) atomicAdd(&pcnt[p], 1.f);
}

__global__ __launch_bounds__(BLK)
void seg_mean_kernel(float* __restrict__ psum, const float* __restrict__ pcnt,
                     int N, int C)
{
  size_t i = (size_t)blockIdx.x * BLK + threadIdx.x;
  if (i >= (size_t)N * C) return;
  psum[i] /= fmaxf(pcnt[i / C], 1.f);
}

__global__ __launch_bounds__(BLK)
void seg_scatter_kernel(float* __restrict__ F, const float* __restrict__ H,
                        const int* __restrict__ seg, int N, int C)
{
  size_t i = (size_t)blockIdx.x * BLK + threadIdx.x;
  if (i >= (size_t)N * C) return;
  int r = (int)(i / C), c = (int)(i % C);
  F[i] += H[(size_t)seg[r] * C + c];
}

// km8[k][i] = (kidx[i]==k) ? cidx[i] : -1   (transposed-conv as gather-conv)
__global__ __launch_bounds__(BLK)
void build_upmap_kernel(const int* __restrict__ kidx, const int* __restrict__ cidx,
                        int N, int* __restrict__ km)
{
  size_t i = (size_t)blockIdx.x * BLK + threadIdx.x;
  if (i >= (size_t)8 * N) return;
  int k = (int)(i / N), r = (int)(i % N);
  km[i] = (kidx[r] == k) ? cidx[r] : -1;
}

__global__ __launch_bounds__(BLK)
void concat_kernel(const float* __restrict__ A, int Ca,
                   const float* __restrict__ B, int Cb,
                   float* __restrict__ O, int N)
{
  int C = Ca + Cb;
  size_t i = (size_t)blockIdx.x * BLK + threadIdx.x;
  if (i >= (size_t)N * C) return;
  int r = (int)(i / C), c = (int)(i % C);
  O[i] = (c < Ca) ? A[(size_t)r * Ca + c] : B[(size_t)r * Cb + (c - Ca)];
}

// ============================================================================
// Host-side orchestration
// ============================================================================
struct Cp { const float *W, *g, *b; };
struct Pp { const float *W, *bL, *g, *b; };
struct Rp { Cp c1, c2; const float *dW, *dg, *db; };
struct StageP { Cp down; Pp pl[2]; Rp res[2]; };
struct UpP { Cp up; Rp res[2]; };

struct Alloc {
  char* base; size_t off; size_t cap;
  void* get(size_t bytes) {
    void* p = base + off;
    off = (off + bytes + 255) & ~(size_t)255;
    return p;
  }
  float* f(size_t n) { return (float*)get(n * sizeof(float)); }
  int*   i(size_t n) { return (int*)get(n * sizeof(int)); }
};

static inline unsigned g1(size_t n) {
  size_t b = (n + BLK - 1) / BLK;
  return b ? (unsigned)b : 1u;
}

static void zero(hipStream_t st, float* p, size_t n) {
  zero_kernel<<<g1(n), BLK, 0, st>>>(p, n);
}

// Nf = rows of F (gather source height)
static void run_gemm(hipStream_t st, Alloc& A, const float* F, int Nf, int Cin,
                     const float* W, int K, const int* kmap, int Nout, int Cout,
                     const float* bias, float* Out)
{
  int Cpad = (Cin + 31) & ~31;
  int CT = Cpad >> 5;
  int ctiles = (Cout + 15) / 16;
  int nw = ctiles < NWMAX ? ctiles : NWMAX;
  int NTp = ((ctiles + nw - 1) / nw) * nw;

  unsigned short* Xh = (unsigned short*)A.get((size_t)Nf * Cpad * 2);
  xtobf_kernel<<<g1((size_t)Nf * Cpad), BLK, 0, st>>>(F, Xh, Nf, Cin, Cpad);

  size_t wfrag = (size_t)K * NTp * CT * 512;
  unsigned short* Wf = (unsigned short*)A.get(wfrag * 2);
  wtobf_kernel<<<g1(wfrag), BLK, 0, st>>>(W, Wf, K, Cin, Cout, CT, NTp);

  dim3 grid((Nout + 15) / 16, NTp / nw);
  wmma_gemm_gather<<<grid, dim3(32 * nw), 0, st>>>(Xh, Cpad, Wf, K, kmap, Nout,
                                                   Cout, bias, Out, CT, NTp);
}

static void run_bn(hipStream_t st, float* X, int N, int C, const float* g,
                   const float* b, const float* mask, const float* res,
                   int relu, float* stat)
{
  colstat_kernel<<<C, BLK, 0, st>>>(X, N, C, mask, stat);
  bn_apply_kernel<<<g1((size_t)N * C), BLK, 0, st>>>(X, X, N, C, stat, g, b,
                                                     res, relu);
}

static void cbr(hipStream_t st, Alloc& A, const float* F, int Nf, int Cin,
                const Cp& p, int K, const int* kmap, int Nout, int Cout,
                float* Out, float* stat)
{
  run_gemm(st, A, F, Nf, Cin, p.W, K, kmap, Nout, Cout, nullptr, Out);
  run_bn(st, Out, Nout, Cout, p.g, p.b, nullptr, nullptr, 1, stat);
}

static void resblock(hipStream_t st, Alloc& A, const float* fin, int N, int Cin,
                     const int* km27, const Rp& p, int Cout, float* out,
                     float* stat)
{
  float* h1 = A.f((size_t)N * Cout);
  run_gemm(st, A, fin, N, Cin, p.c1.W, 27, km27, N, Cout, nullptr, h1);
  run_bn(st, h1, N, Cout, p.c1.g, p.c1.b, nullptr, nullptr, 1, stat);
  run_gemm(st, A, h1, N, Cout, p.c2.W, 27, km27, N, Cout, nullptr, out); // raw h2
  const float* s = fin;
  if (p.dW) {
    float* sbuf = A.f((size_t)N * Cout);
    run_gemm(st, A, fin, N, Cin, p.dW, 1, nullptr, N, Cout, nullptr, sbuf);
    run_bn(st, sbuf, N, Cout, p.dg, p.db, nullptr, nullptr, 0, stat);
    s = sbuf;
  }
  // out = relu(bn(h2raw) + s)
  run_bn(st, out, N, Cout, p.c2.g, p.c2.b, nullptr, s, 1, stat);
}

static void pillar_run(hipStream_t st, Alloc& A, float* f, int N, int C,
                       const int* seg, const Pp* pl, float* stat)
{
  float* psum = A.f((size_t)N * C);
  float* pcnt = A.f((size_t)N);
  float* h1   = A.f((size_t)N * C);
  size_t total = (size_t)N * C;
  zero(st, psum, total);
  zero(st, pcnt, (size_t)N);
  seg_accum_kernel<<<g1(total), BLK, 0, st>>>(f, seg, N, C, psum, pcnt);
  seg_mean_kernel<<<g1(total), BLK, 0, st>>>(psum, pcnt, N, C);
  run_gemm(st, A, psum, N, C, pl[0].W, 1, nullptr, N, C, pl[0].bL, h1);
  run_bn(st, h1, N, C, pl[0].g, pl[0].b, pcnt, nullptr, 1, stat);
  run_gemm(st, A, h1, N, C, pl[1].W, 1, nullptr, N, C, pl[1].bL, psum);
  run_bn(st, psum, N, C, pl[1].g, pl[1].b, pcnt, nullptr, 1, stat);
  seg_scatter_kernel<<<g1(total), BLK, 0, st>>>(f, psum, seg, N, C);
}

static void stage_run(hipStream_t st, Alloc& A, const float* fin, int Nprev,
                      int Cin, const int* dmap, const int* km27, const int* seg,
                      int Nl, const StageP& S, int Cout, float* xout, float* stat)
{
  float* cur = A.f((size_t)Nl * Cin);
  cbr(st, A, fin, Nprev, Cin, S.down, 8, dmap, Nl, Cin, cur, stat);
  pillar_run(st, A, cur, Nl, Cin, seg, S.pl, stat);
  float* r0 = A.f((size_t)Nl * Cout);
  resblock(st, A, cur, Nl, Cin, km27, S.res[0], Cout, r0, stat);
  resblock(st, A, r0, Nl, Cout, km27, S.res[1], Cout, xout, stat);
}

static void up_run(hipStream_t st, Alloc& A, const float* fc, int Nc, int Cc,
                   const float* skip, int Cskip, const int* kidx,
                   const int* cidx, int Nf, const int* km27, const UpP& U,
                   int Cup, int Cout, float* yout, float* stat)
{
  int* km8 = A.i((size_t)8 * Nf);
  build_upmap_kernel<<<g1((size_t)8 * Nf), BLK, 0, st>>>(kidx, cidx, Nf, km8);
  float* u = A.f((size_t)Nf * Cup);
  run_gemm(st, A, fc, Nc, Cc, U.up.W, 8, km8, Nf, Cup, nullptr, u);
  run_bn(st, u, Nf, Cup, U.up.g, U.up.b, nullptr, nullptr, 1, stat);
  int Ccat = Cup + Cskip;
  float* cat = A.f((size_t)Nf * Ccat);
  concat_kernel<<<g1((size_t)Nf * Ccat), BLK, 0, st>>>(u, Cup, skip, Cskip,
                                                       cat, Nf);
  float* r0 = A.f((size_t)Nf * Cout);
  resblock(st, A, cat, Nf, Ccat, km27, U.res[0], Cout, r0, stat);
  resblock(st, A, r0, Nf, Cout, km27, U.res[1], Cout, yout, stat);
}

// ============================================================================
// kernel_launch: parse d_in (setup_inputs dict order, params flattened
// depth-first in insertion order), run the full forward into d_out (f32).
// ============================================================================
extern "C" void kernel_launch(void* const* d_in, const int* in_sizes, int n_in,
                              void* d_out, int out_size, void* d_ws,
                              size_t ws_size, hipStream_t stream)
{
  (void)n_in; (void)out_size;
  const float* feats = (const float*)d_in[1];

  int p = 2;  // params start after coords, feats
  auto PF = [&](void) { return (const float*)d_in[p++]; };
  auto getCp = [&](Cp& c) { c.W = PF(); c.g = PF(); c.b = PF(); };
  auto getPp = [&](Pp& q) { q.W = PF(); q.bL = PF(); q.g = PF(); q.b = PF(); };
  auto getRp = [&](Rp& r, bool dw) {
    getCp(r.c1); getCp(r.c2);
    if (dw) { r.dW = PF(); r.dg = PF(); r.db = PF(); }
    else    { r.dW = nullptr; r.dg = nullptr; r.db = nullptr; }
  };

  Cp stem0, stem1; getCp(stem0); getCp(stem1);
  StageP st1, st2, st3, st4;
  // stage1: res0 32->32 (no dW); stages 2-4: res0 changes channels (dW)
  getCp(st1.down); getPp(st1.pl[0]); getPp(st1.pl[1]);
  getRp(st1.res[0], false); getRp(st1.res[1], false);
  getCp(st2.down); getPp(st2.pl[0]); getPp(st2.pl[1]);
  getRp(st2.res[0], true);  getRp(st2.res[1], false);
  getCp(st3.down); getPp(st3.pl[0]); getPp(st3.pl[1]);
  getRp(st3.res[0], true);  getRp(st3.res[1], false);
  getCp(st4.down); getPp(st4.pl[0]); getPp(st4.pl[1]);
  getRp(st4.res[0], true);  getRp(st4.res[1], false);
  UpP up1, up2, up3, up4;
  getCp(up1.up); getRp(up1.res[0], true); getRp(up1.res[1], false);
  getCp(up2.up); getRp(up2.res[0], true); getRp(up2.res[1], false);
  getCp(up3.up); getRp(up3.res[0], true); getRp(up3.res[1], false);
  getCp(up4.up); getRp(up4.res[0], true); getRp(up4.res[1], false);
  const float* clsW = PF();
  const float* clsB = PF();

  // integer side-inputs (p should be 183 here)
  const int* k0m = (const int*)d_in[p++];
  const int* k1m = (const int*)d_in[p++];
  const int* k2m = (const int*)d_in[p++];
  const int* k3m = (const int*)d_in[p++];
  const int* k4m = (const int*)d_in[p++];
  const int* d1m = (const int*)d_in[p++];
  const int* d2m = (const int*)d_in[p++];
  const int* d3m = (const int*)d_in[p++];
  const int* d4m = (const int*)d_in[p++];
  const int* u1k = (const int*)d_in[p++]; const int* u1c = (const int*)d_in[p++];
  const int* u2k = (const int*)d_in[p++]; const int* u2c = (const int*)d_in[p++];
  const int* u3k = (const int*)d_in[p++]; const int* u3c = (const int*)d_in[p++];
  const int* u4k = (const int*)d_in[p++]; const int* u4c = (const int*)d_in[p++];
  int segBase = p;
  const int* seg1 = (const int*)d_in[p++];
  const int* seg2 = (const int*)d_in[p++];
  const int* seg3 = (const int*)d_in[p++];
  const int* seg4 = (const int*)d_in[p++];

  const int N0 = in_sizes[0] / 4;            // coords [N0,4]
  const int N1 = in_sizes[segBase + 0];
  const int N2 = in_sizes[segBase + 1];
  const int N3 = in_sizes[segBase + 2];
  const int N4 = in_sizes[segBase + 3];

  Alloc A{(char*)d_ws, 0, ws_size};
  float* stat = A.f(1024);                   // stats scratch (2*Cmax+1 <= 769)
  float* x0 = A.f((size_t)N0 * 32);
  float* x1 = A.f((size_t)N1 * 32);
  float* x2 = A.f((size_t)N2 * 64);
  float* x3 = A.f((size_t)N3 * 128);
  float* x4 = A.f((size_t)N4 * 256);
  float* y1 = A.f((size_t)N3 * 256);
  float* y2 = A.f((size_t)N2 * 128);
  float* y3 = A.f((size_t)N1 * 96);
  float* y4 = A.f((size_t)N0 * 96);
  const size_t tempBase = A.off;

  // ---- stem ----
  {
    float* t1 = A.f((size_t)N0 * 32);
    cbr(stream, A, feats, N0, 3, stem0, 27, k0m, N0, 32, t1, stat);
    cbr(stream, A, t1, N0, 32, stem1, 27, k0m, N0, 32, x0, stat);
    A.off = tempBase;
  }
  // ---- encoder ----
  stage_run(stream, A, x0, N0, 32,  d1m, k1m, seg1, N1, st1, 32,  x1, stat); A.off = tempBase;
  stage_run(stream, A, x1, N1, 32,  d2m, k2m, seg2, N2, st2, 64,  x2, stat); A.off = tempBase;
  stage_run(stream, A, x2, N2, 64,  d3m, k3m, seg3, N3, st3, 128, x3, stat); A.off = tempBase;
  stage_run(stream, A, x3, N3, 128, d4m, k4m, seg4, N4, st4, 256, x4, stat); A.off = tempBase;
  // ---- decoder ----
  up_run(stream, A, x4, N4, 256, x3, 128, u1k, u1c, N3, k3m, up1, 256, 256, y1, stat); A.off = tempBase;
  up_run(stream, A, y1, N3, 256, x2, 64,  u2k, u2c, N2, k2m, up2, 128, 128, y2, stat); A.off = tempBase;
  up_run(stream, A, y2, N2, 128, x1, 32,  u3k, u3c, N1, k1m, up3, 96,  96,  y3, stat); A.off = tempBase;
  up_run(stream, A, y3, N1, 96,  x0, 32,  u4k, u4c, N0, k0m, up4, 96,  96,  y4, stat); A.off = tempBase;
  // ---- classifier: d_out = y4 @ W + b  (f32, [N0,19]) ----
  run_gemm(stream, A, y4, N0, 96, clsW, 1, nullptr, N0, 19, clsB, (float*)d_out);
}
